// inputToLeafSinkhorn_85547158602257
// MI455X (gfx1250) — matrix operand order repaired and checked by reading, not compiled, not run on hardware
//
#include <hip/hip_runtime.h>
#include <hip/hip_bf16.h>
#include <stdint.h>

// ---------------------------------------------------------------------------
// Problem constants (from reference)
// ---------------------------------------------------------------------------
#define BATCH       65536
#define NUM_IN      1024
#define NUM_LEAVES  1024
#define SINK_ITERS  20

typedef __attribute__((ext_vector_type(16))) __bf16       v16bf;
typedef __attribute__((ext_vector_type(8)))  float        v8f;
typedef __attribute__((ext_vector_type(4)))  unsigned int v4u;
typedef __attribute__((ext_vector_type(8)))  int          v8i;
typedef __attribute__((ext_vector_type(4)))  int          v4i;

union FragBF {
    uint4  q[2];   // q[0] = bytes 0..15 (elements 0..7), q[1] = bytes 16..31
    v16bf  v;
};

#if __has_builtin(__builtin_amdgcn_tensor_load_to_lds)
#define USE_TDM 1
#else
#define USE_TDM 0
#endif

__device__ __forceinline__ void wait_tensorcnt0() {
#if __has_builtin(__builtin_amdgcn_s_wait_tensorcnt)
    __builtin_amdgcn_s_wait_tensorcnt(0);
#else
    asm volatile("s_wait_tensorcnt 0x0" ::: "memory");
#endif
}
__device__ __forceinline__ void wait_tensorcnt1() {
#if __has_builtin(__builtin_amdgcn_s_wait_tensorcnt)
    __builtin_amdgcn_s_wait_tensorcnt(1);
#else
    asm volatile("s_wait_tensorcnt 0x1" ::: "memory");
#endif
}

// ---------------------------------------------------------------------------
// 1) A0 = exp(clip((logits + noise)/T, -10, 10))
// ---------------------------------------------------------------------------
__global__ void k_init(const float* __restrict__ logits,
                       const float* __restrict__ noise,
                       float* __restrict__ A) {
    int i = blockIdx.x * blockDim.x + threadIdx.x;
    float la = (logits[i] + noise[i]) * (1.0f / 3.0f);
    la = fminf(fmaxf(la, -10.0f), 10.0f);
    A[i] = __expf(la);
}

// ---------------------------------------------------------------------------
// 2a) Row normalize: one block per row, block reduction then divide
// ---------------------------------------------------------------------------
__global__ void k_row_norm(float* __restrict__ A) {
    __shared__ float red[256];
    const int row = blockIdx.x;
    float* rp = A + (size_t)row * NUM_IN;

    float s = 0.0f;
    for (int c = threadIdx.x; c < NUM_IN; c += 256) s += rp[c];
    red[threadIdx.x] = s;
    __syncthreads();
    for (int off = 128; off > 0; off >>= 1) {
        if (threadIdx.x < off) red[threadIdx.x] += red[threadIdx.x + off];
        __syncthreads();
    }
    const float rsum = red[0];
    for (int c = threadIdx.x; c < NUM_IN; c += 256) rp[c] = rp[c] / rsum;
}

// ---------------------------------------------------------------------------
// 2b) Column sums (deterministic serial-per-column, coalesced across lanes)
// ---------------------------------------------------------------------------
__global__ void k_col_sum(const float* __restrict__ A, float* __restrict__ colsum) {
    const int c = blockIdx.x * blockDim.x + threadIdx.x;  // 0..1023
    float s = 0.0f;
#pragma unroll 4
    for (int r = 0; r < NUM_LEAVES; ++r) s += A[(size_t)r * NUM_IN + c];
    colsum[c] = s;
}

__global__ void k_col_scale(float* __restrict__ A, const float* __restrict__ colsum) {
    const int i = blockIdx.x * 256 + threadIdx.x;
    const int c = i & (NUM_IN - 1);
    A[i] = A[i] / colsum[c];
}

// ---------------------------------------------------------------------------
// 3) nan_to_num + convert to bf16 (P stays row-major [leaf][input] — this is
//    exactly the K-contiguous B-operand layout for the GEMM).
// ---------------------------------------------------------------------------
__global__ void k_to_bf16(const float* __restrict__ A, __bf16* __restrict__ Pbf) {
    const int i = blockIdx.x * 256 + threadIdx.x;
    float v = A[i];
    if (v != v) v = 0.0f;                 // nan -> 0
    else if (v > 3.0e38f) v = 1.0f;       // +inf -> 1
    else if (v < -3.0e38f) v = 0.0f;      // -inf -> 0
    Pbf[i] = (__bf16)v;
}

// ---------------------------------------------------------------------------
// 4) GEMM: out[M=65536, N=1024] = X[M, K=1024] * P^T   (B[k][n] = P[n][k])
//    BM=128, BN=128, BK=64, 256 threads = 8 waves (wave32), each wave 32x64.
//    bf16 WMMA 16x16x32 with f32 accumulation.
//    B tiles are staged by the Tensor Data Mover (double-buffered, TENSORcnt
//    pipelined); A tiles go through VALU (f32 -> bf16 conversion needed).
// ---------------------------------------------------------------------------
#define BM   128
#define BN   128
#define BK   64
#define NK   (NUM_IN / BK)   // 16 K-steps
#define LDSK 72   // padded row (bf16): 144B stride = 32 dwords + 4 pad dwords

#if USE_TDM
// Issue one TDM DMA: 128x64 bf16 tile of Pbf (row nBlock, col k) -> LDS,
// with 4 pad DWORDs inserted after every 32 DWORDs (row stride 144B = LDSK).
__device__ __forceinline__ void tdm_load_B(const __bf16* Pbf, int nBlock, int k,
                                           unsigned lds_byte_off) {
    const unsigned long long gaddr =
        (unsigned long long)(Pbf + (size_t)nBlock * NUM_IN + k);
    // ---- D# group 0: count=1 | lds_addr | global_addr | type=2
    v4u g0;
    g0.x = 1u;                                   // count=1, no gather
    g0.y = lds_byte_off;                         // bits 63:32 lds_addr
    g0.z = (unsigned)gaddr;                      // global_addr[31:0]
    g0.w = (unsigned)((gaddr >> 32) & 0x1FFFFFFu) | (2u << 30);  // [120:96]|type=2
    // ---- D# group 1
    v8i g1;
    g1[0] = (int)((1u << 16)      // data_size = 1 -> 2 bytes
                | (1u << 20)      // pad_enable
                | (4u << 22)      // pad_interval: 4 -> every 32 DWORDs (=128B row)
                | (3u << 25));    // pad_amount: 3 -> 4 DWORDs (=16B pad)
    g1[1] = (int)(1024u << 16);   // tensor_dim0[15:0]=1024 in bits 63:48
    g1[2] = (int)(1024u << 16);   // dim0[31:16]=0 | tensor_dim1[15:0]=1024
    g1[3] = (int)(64u << 16);     // dim1[31:16]=0 | tile_dim0=64
    g1[4] = 128;                  // tile_dim1=128 | tile_dim2=0
    g1[5] = 1024;                 // tensor_dim0_stride[31:0]
    g1[6] = 0;                    // stride0[47:32]=0 | dim1_stride[15:0]=0
    g1[7] = 16;                   // tensor_dim1_stride[47:16] = (1<<20)>>16
    v4i g2 = {0, 0, 0, 0};        // unused higher dims
    v4i g3 = {0, 0, 0, 0};
#if defined(__clang_major__) && (__clang_major__ >= 23)
    v8i g4 = {0, 0, 0, 0, 0, 0, 0, 0};
    __builtin_amdgcn_tensor_load_to_lds(g0, g1, g2, g3, g4, 0);
#else
    __builtin_amdgcn_tensor_load_to_lds(g0, g1, g2, g3, 0);
#endif
}
#endif

__global__ __launch_bounds__(256)
void k_gemm(const float* __restrict__ X,
            const __bf16* __restrict__ Pbf,
            float* __restrict__ Out) {
    __shared__ __align__(16) __bf16 sA[BM * LDSK];
    __shared__ __align__(16) __bf16 sB[2][BN * LDSK];

    const int tid  = threadIdx.x;
    const int lane = tid & 31;
    const int wave = tid >> 5;
    const int l    = lane & 15;   // N / M index inside 16-wide group
    const int half = lane >> 4;   // half-wave selector

    const int nBlock = blockIdx.x * BN;   // 8 N-tiles (inner grid dim -> x reuse in L2)
    const int mBlock = blockIdx.y * BM;   // 512 M-tiles

    const int wm = (wave & 3) * 32;       // wave's row base inside block tile
    const int wn = (wave >> 2) * 64;      // wave's col base inside block tile

    v8f acc[2][4];
#pragma unroll
    for (int mt = 0; mt < 2; ++mt)
#pragma unroll
        for (int nt = 0; nt < 4; ++nt)
#pragma unroll
            for (int e = 0; e < 8; ++e) acc[mt][nt][e] = 0.0f;

#if USE_TDM
    // Byte offsets of the two B buffers inside LDS (flat addr low 32 bits).
    const unsigned sB_off0 = (unsigned)(unsigned long long)(void*)&sB[0][0];
    const unsigned sB_off1 = (unsigned)(unsigned long long)(void*)&sB[1][0];
    if (wave == 0) tdm_load_B(Pbf, nBlock, 0, sB_off0);   // prime buffer 0
#endif

    for (int i = 0; i < NK; ++i) {
        const int k0 = i * BK;
        __syncthreads();   // all reads of sA / sB[(i+1)&1] from prior iters done

#if USE_TDM
        // Kick DMA for the *next* K-step into the other buffer; it overlaps
        // the A staging below and the entire compute phase of this step.
        if (wave == 0 && (i + 1) < NK)
            tdm_load_B(Pbf, nBlock, k0 + BK, ((i + 1) & 1) ? sB_off1 : sB_off0);
#endif

        // ---- stage A tile: 128x64 f32 -> bf16 into LDS (8 float4 per thread)
#pragma unroll
        for (int it = 0; it < 8; ++it) {
            const int idx = it * 256 + tid;        // 0..2047 vec4-slots
            const int row = idx >> 4;              // 16 float4 per row
            const int c4  = (idx & 15) << 2;       // float offset in row
            const float* gp = X + (size_t)(mBlock + row) * NUM_IN + k0 + c4;
            const float4 vv = *reinterpret_cast<const float4*>(gp);
            if (k0 + BK < NUM_IN) __builtin_prefetch(gp + BK, 0, 1);
            union { __bf16 h[4]; uint2 u; } pk;
            pk.h[0] = (__bf16)vv.x; pk.h[1] = (__bf16)vv.y;
            pk.h[2] = (__bf16)vv.z; pk.h[3] = (__bf16)vv.w;
            *reinterpret_cast<uint2*>(sA + row * LDSK + c4) = pk.u;
        }

#if USE_TDM
        // Allow one outstanding DMA (the one for step i+1); require the DMA
        // for *this* step's buffer to be complete. In-order TENSORcnt.
        if (wave == 0) {
            if ((i + 1) < NK) wait_tensorcnt1();
            else              wait_tensorcnt0();
        }
        const __bf16* __restrict__ sBc = sB[i & 1];
#else
        // Fallback: synchronous copy of B tile (4 x 16B per thread)
        __bf16* sBw = sB[i & 1];
#pragma unroll
        for (int it = 0; it < 4; ++it) {
            const int idx = it * 256 + tid;        // 0..1023 16B-slots
            const int row = idx >> 3;              // 8 x 16B per row
            const int c8  = (idx & 7) << 3;        // bf16 offset in row
            const uint4 vv = *reinterpret_cast<const uint4*>(
                Pbf + (size_t)(nBlock + row) * NUM_IN + k0 + c8);
            *reinterpret_cast<uint4*>(sBw + row * LDSK + c8) = vv;
        }
        const __bf16* __restrict__ sBc = sB[i & 1];
#endif
        __syncthreads();

        // ---- compute: two K sub-steps of 32
#pragma unroll
        for (int ks = 0; ks < BK; ks += 32) {
            FragBF a[2];
#pragma unroll
            for (int mt = 0; mt < 2; ++mt) {
                // A 16x32 layout: half 0 holds K {ks+0..7 | ks+16..23},
                //                 half 1 holds K {ks+8..15 | ks+24..31}
                const __bf16* base = sA + (wm + mt * 16 + l) * LDSK + ks + half * 8;
                a[mt].q[0] = *reinterpret_cast<const uint4*>(base);
                a[mt].q[1] = *reinterpret_cast<const uint4*>(base + 16);
            }
            FragBF b[4];
#pragma unroll
            for (int nt = 0; nt < 4; ++nt) {
                // B 32x16 layout: half 0 holds K {ks+0..7 | ks+16..23},
                //                 half 1 holds K {ks+8..15 | ks+24..31}
                const __bf16* base = sBc + (wn + nt * 16 + l) * LDSK + ks + half * 8;
                b[nt].q[0] = *reinterpret_cast<const uint4*>(base);
                b[nt].q[1] = *reinterpret_cast<const uint4*>(base + 16);
            }
#pragma unroll
            for (int mt = 0; mt < 2; ++mt)
#pragma unroll
                for (int nt = 0; nt < 4; ++nt)
                    acc[mt][nt] = __builtin_amdgcn_wmma_f32_16x16x32_bf16(
                        false, a[mt].v, false, b[nt].v,
                        (short)0, acc[mt][nt], false, false);
        }
    }

    // ---- write back: C/D layout — lane%16 = N, VGPR e = M row (+8 for half 1)
#pragma unroll
    for (int mt = 0; mt < 2; ++mt) {
#pragma unroll
        for (int nt = 0; nt < 4; ++nt) {
            const int col     = nBlock + wn + nt * 16 + l;
            const int rowBase = mBlock + wm + mt * 16 + half * 8;
#pragma unroll
            for (int e = 0; e < 8; ++e)
                Out[(size_t)(rowBase + e) * NUM_LEAVES + col] = acc[mt][nt][e];
        }
    }
}

// ---------------------------------------------------------------------------
// Launch
// ---------------------------------------------------------------------------
extern "C" void kernel_launch(void* const* d_in, const int* in_sizes, int n_in,
                              void* d_out, int out_size, void* d_ws, size_t ws_size,
                              hipStream_t stream) {
    const float* x      = (const float*)d_in[0];   // [65536, 1024]
    const float* logits = (const float*)d_in[1];   // [1024, 1024]
    const float* noise  = (const float*)d_in[2];   // [1024, 1024]
    float* out = (float*)d_out;                    // [65536, 1024]

    // Workspace layout: A (f32, 4MB) | colsum (f32, 4KB) | Pbf (bf16, 2MB)
    float*  A      = (float*)d_ws;
    float*  colsum = A + (size_t)NUM_LEAVES * NUM_IN;
    __bf16* Pbf    = (__bf16*)(colsum + NUM_IN);

    const int elems  = NUM_LEAVES * NUM_IN;        // 1M
    const int eb     = elems / 256;                // 4096 blocks

    k_init<<<eb, 256, 0, stream>>>(logits, noise, A);
    for (int it = 0; it < SINK_ITERS; ++it) {
        k_row_norm <<<NUM_LEAVES, 256, 0, stream>>>(A);
        k_col_sum  <<<NUM_IN / 256, 256, 0, stream>>>(A, colsum);
        k_col_scale<<<eb, 256, 0, stream>>>(A, colsum);
    }
    k_to_bf16<<<eb, 256, 0, stream>>>(A, Pbf);

    dim3 grid(NUM_LEAVES / BN, BATCH / BM);        // (8, 512)
    k_gemm<<<grid, 256, 0, stream>>>(x, Pbf, out);
}
